// ChemGeomFeatEncoder_79173427135041
// MI455X (gfx1250) — matrix operand order repaired and checked by reading, not compiled, not run on hardware
//
#include <hip/hip_runtime.h>

// ---------------------------------------------------------------------------
// ChemGeomFeatEncoder for MI455X (gfx1250): bf16 WMMA GEMM pipeline.
// All linear layers are followed by train-mode BatchNorm -> biases cancel and
// are skipped. BN stats accumulated in the GEMM epilogue (register-reduced ->
// LDS f32 atomics -> global f32 atomics), finalized to per-column scale/shift,
// and fused (with SiLU) into the next GEMM's A staging.
// Weights are pre-packed once into WMMA B-fragment layout so B operands are
// loaded straight from global memory as 2x uint4 per fragment (no LDS).
// All GEMMs have K % 32 == 0 (the K=144 edge-feature matrix is padded to 160).
// ---------------------------------------------------------------------------

typedef __attribute__((ext_vector_type(16))) __bf16 v16bf;
typedef __attribute__((ext_vector_type(8)))  float  v8f;

union FragAB {
    v16bf v;
    unsigned int u[8];
    uint4 q[2];
};
static_assert(sizeof(FragAB) == 32, "frag size");

__device__ __forceinline__ unsigned short f2bf(float x) {
    unsigned int u = __float_as_uint(x);
    unsigned int r = (u + 0x7FFFu + ((u >> 16) & 1u)) >> 16;  // RNE
    return (unsigned short)r;
}
__device__ __forceinline__ float bf2f(unsigned short s) {
    return __uint_as_float(((unsigned int)s) << 16);
}
__device__ __forceinline__ float bf_lo(unsigned int d) {
    return __uint_as_float(d << 16);
}
__device__ __forceinline__ float bf_hi(unsigned int d) {
    return __uint_as_float(d & 0xffff0000u);
}

// ---------------------------------------------------------------------------
// GEMM: Z[M,N](bf16) = act(bn(A))[M,K] @ W[K,N]; f32 accumulate via
// v_wmma_f32_16x16x32_bf16. Block = 256 threads = 8 waves (4 M-groups x
// 2 N-groups), tile 128(M) x 64(N), K step 32 (K % 32 == 0 required).
// BN_SILU=true fuses y = silu(x*scale + shift) into A staging.
// All index products fit in i32 for this problem (max ~2.05e8).
// ---------------------------------------------------------------------------
#define TM 128
#define TN 64
#define TK 32

template <bool BN_SILU>
__global__ __launch_bounds__(256, 2)
void gemm_bf16_bn(const unsigned short* __restrict__ A,     // [M,K] bf16
                  const float* __restrict__ a_scale,        // [K] (BN_SILU only)
                  const float* __restrict__ a_shift,        // [K] (BN_SILU only)
                  const unsigned short* __restrict__ Wf,    // packed [N/16][kt][32][16]
                  unsigned short* __restrict__ Z,           // [M,N] bf16
                  float* __restrict__ colsum,               // [N]
                  float* __restrict__ colsq,                // [N]
                  int M, int K, int N)
{
    __shared__ alignas(16) unsigned short As[TM][TK];  // 8 KB
    __shared__ float csum[TN];
    __shared__ float csq[TN];

    const int tid  = threadIdx.x;
    const int lane = tid & 31;
    const int wave = tid >> 5;
    const int wm   = wave & 3;   // M-group (32 rows each)
    const int wn   = wave >> 2;  // N-group (32 cols each)
    const int m0   = blockIdx.x * TM;
    const int n0   = blockIdx.y * TN;
    const int ktiles = K / TK;

    if (tid < TN) { csum[tid] = 0.0f; csq[tid] = 0.0f; }

    v8f zero = {0.f, 0.f, 0.f, 0.f, 0.f, 0.f, 0.f, 0.f};
    v8f acc[2][2] = {{zero, zero}, {zero, zero}};

    // A staging: thread -> (row, 16-col half); 32 B chunk, 16 B-aligned.
    // Rows beyond M are clamped to row M-1: they only feed C rows that are
    // masked out of both the store and the BN statistics.
    const int ar = tid >> 1;
    const int ah = (tid & 1) * 16;
    int am = m0 + ar;
    if (am >= M) am = M - 1;
    const unsigned short* arow = A + am * K;

    for (int kb = 0; kb < ktiles; ++kb) {
        const int k0 = kb * TK;
        __syncthreads();

        // ---------------- stage A tile ----------------
        union { uint4 v[2]; unsigned int u[8]; } buf;
        {
            const uint4* p = (const uint4*)(arow + k0 + ah);
            buf.v[0] = p[0];
            buf.v[1] = p[1];
        }
        __builtin_prefetch(arow + k0 + TK + ah, 0, 1);
        if (BN_SILU) {
            const float* scp = a_scale + k0 + ah;
            const float* shp = a_shift + k0 + ah;
#pragma unroll
            for (int q = 0; q < 8; ++q) {
                float xl = bf_lo(buf.u[q]) * scp[2 * q] + shp[2 * q];
                float xh = bf_hi(buf.u[q]) * scp[2 * q + 1] + shp[2 * q + 1];
                xl = xl / (1.0f + __expf(-xl));
                xh = xh / (1.0f + __expf(-xh));
                buf.u[q] = (unsigned int)f2bf(xl) |
                           ((unsigned int)f2bf(xh) << 16);
            }
        }
        *((uint4*)&As[ar][ah])     = buf.v[0];
        *((uint4*)&As[ar][ah + 8]) = buf.v[1];
        __syncthreads();

        // ---------------- fragments ----------------
        FragAB aF[2], bF[2];
        const int kA = (lane >= 16) ? 8 : 0;  // A layout: hi lanes hold K 8-15/24-31
#pragma unroll
        for (int mi = 0; mi < 2; ++mi) {
            const int r = wm * 32 + mi * 16 + (lane & 15);
            aF[mi].q[0] = *(const uint4*)&As[r][kA];       // K 0..7   (pairs)
            aF[mi].q[1] = *(const uint4*)&As[r][16 + kA];  // K 16..23 (pairs)
        }
#pragma unroll
        for (int ni = 0; ni < 2; ++ni) {
            const int nb = (n0 >> 4) + wn * 2 + ni;
            const uint4* p = (const uint4*)(
                Wf + ((nb * ktiles + kb) * 32 + lane) * 16);
            bF[ni].q[0] = p[0];
            bF[ni].q[1] = p[1];
        }

        acc[0][0] = __builtin_amdgcn_wmma_f32_16x16x32_bf16(
            false, aF[0].v, false, bF[0].v, (short)0, acc[0][0], false, false);
        acc[0][1] = __builtin_amdgcn_wmma_f32_16x16x32_bf16(
            false, aF[0].v, false, bF[1].v, (short)0, acc[0][1], false, false);
        acc[1][0] = __builtin_amdgcn_wmma_f32_16x16x32_bf16(
            false, aF[1].v, false, bF[0].v, (short)0, acc[1][0], false, false);
        acc[1][1] = __builtin_amdgcn_wmma_f32_16x16x32_bf16(
            false, aF[1].v, false, bF[1].v, (short)0, acc[1][1], false, false);
    }

    // ---------------- epilogue ----------------
    const int rofs = (lane >= 16) ? 8 : 0;  // C/D: VGPR r -> row r or r+8
    if (m0 + TM <= M) {
        // Full tile (the common case; every tile for the edge GEMMs).
#pragma unroll
        for (int ni = 0; ni < 2; ++ni) {
            const int nloc = wn * 32 + ni * 16 + (lane & 15);
            const int n = n0 + nloc;
            float s = 0.0f, q = 0.0f;
#pragma unroll
            for (int mi = 0; mi < 2; ++mi) {
                const int mb = m0 + wm * 32 + mi * 16 + rofs;
#pragma unroll
                for (int rr = 0; rr < 8; ++rr) {
                    const float v = acc[mi][ni][rr];
                    Z[(mb + rr) * N + n] = f2bf(v);
                    s += v;
                    q += v * v;
                }
            }
            atomicAdd(&csum[nloc], s);
            atomicAdd(&csq[nloc], q);
        }
    } else {
        // Partial tile: per-row mask (last block of the vertex/node GEMMs).
#pragma unroll
        for (int ni = 0; ni < 2; ++ni) {
            const int nloc = wn * 32 + ni * 16 + (lane & 15);
            const int n = n0 + nloc;
            float s = 0.0f, q = 0.0f;
#pragma unroll
            for (int mi = 0; mi < 2; ++mi) {
                const int mb = m0 + wm * 32 + mi * 16 + rofs;
#pragma unroll
                for (int rr = 0; rr < 8; ++rr) {
                    const int m = mb + rr;
                    if (m < M) {
                        const float v = acc[mi][ni][rr];
                        Z[m * N + n] = f2bf(v);
                        s += v;
                        q += v * v;
                    }
                }
            }
            atomicAdd(&csum[nloc], s);
            atomicAdd(&csq[nloc], q);
        }
    }
    __syncthreads();
    if (tid < TN) {
        atomicAdd(&colsum[n0 + tid], csum[tid]);
        atomicAdd(&colsq[n0 + tid], csq[tid]);
    }
}

// Pack weights W[K,N] (f32) into WMMA B-fragment layout:
// Wf[nb][kb][lane][el], nb = n/16, kb = k-block of 32; zero-pad k >= K.
__global__ void pack_w(const float* __restrict__ W,
                       unsigned short* __restrict__ Wf,
                       int K, int N, int ktiles)
{
    const long long i = (long long)blockIdx.x * 256 + threadIdx.x;
    const long long total = (long long)(N >> 4) * ktiles * 512;
    if (i >= total) return;
    const int el   = (int)(i & 15);
    const int lane = (int)((i >> 4) & 31);
    long long t = i >> 9;
    const int kb = (int)(t % ktiles);
    const int nb = (int)(t / ktiles);
    const int k0l = (lane >= 16) ? 8 : 0;
    const int k = kb * 32 + k0l + el + ((el >= 8) ? 8 : 0);
    const int n = nb * 16 + (lane & 15);
    Wf[i] = (k < K) ? f2bf(W[(long long)k * N + n]) : (unsigned short)0;
}

// colsum/colsq -> (scale, shift): y = z*scale + shift
__global__ void bn_finalize(const float* __restrict__ sum, const float* __restrict__ sq,
                            const float* __restrict__ gamma, const float* __restrict__ beta,
                            float* __restrict__ scale, float* __restrict__ shift,
                            int N, float invM)
{
    const int i = threadIdx.x;
    if (i < N) {
        const float m = sum[i] * invM;
        const float v = sq[i] * invM - m * m;
        const float s = gamma[i] * rsqrtf(v + 1e-5f);
        scale[i] = s;
        shift[i] = beta[i] - m * s;
    }
}

// f32 -> bf16, 4 elements per thread (n must be a multiple of 4)
__global__ void conv_f32_bf16_v4(const float* __restrict__ src,
                                 unsigned short* __restrict__ dst, long long n4)
{
    const long long i = (long long)blockIdx.x * 256 + threadIdx.x;
    if (i >= n4) return;
    const float4 f = ((const float4*)src)[i];
    uint2 o;
    o.x = (unsigned int)f2bf(f.x) | ((unsigned int)f2bf(f.y) << 16);
    o.y = (unsigned int)f2bf(f.z) | ((unsigned int)f2bf(f.w) << 16);
    ((uint2*)dst)[i] = o;
}

// Build [E,160] bf16 edge features: 128 gathered chem + 8 dist-gdf +
// 8 ang-gdf + 16 zero pad (so the GEMM K is a multiple of 32).
// 4 edges per 256-thread block, 64 threads per edge.
#define XE_LD 160

__global__ __launch_bounds__(256)
void edge_feats(const unsigned short* __restrict__ gx_bf,  // [NN,128] bf16
                const float* __restrict__ node_pos,
                const float* __restrict__ verts,
                const float* __restrict__ vnormals,
                const float* __restrict__ dist,
                const int* __restrict__ ind,
                const int* __restrict__ vids,
                unsigned short* __restrict__ Xe,           // [E,XE_LD]
                int E)
{
    const int e = blockIdx.x * 4 + (threadIdx.x >> 6);
    const int j = threadIdx.x & 63;
    if (e >= E) return;
    const int nd = ind[e];
    unsigned short* row = Xe + (long long)e * XE_LD;  // 320 B rows, 16 B-aligned
    if (j < 16) {  // 128 bf16 = 16 uint4
        ((uint4*)row)[j] = ((const uint4*)(gx_bf + (long long)nd * 128))[j];
    } else if (j < 24) {
        const int jj = j - 16;
        const float d = dist[e];
        const float var = 8.0f / 7.0f;
        const float t = (d - var * (float)jj) / var;   // centers linspace(0,8,8)
        row[128 + jj] = f2bf(__expf(-t * t));
    } else if (j < 32) {
        const int jj = j - 24;
        const int vv = vids[e];
        const float vx = node_pos[nd * 3 + 0] - verts[vv * 3 + 0];
        const float vy = node_pos[nd * 3 + 1] - verts[vv * 3 + 1];
        const float vz = node_pos[nd * 3 + 2] - verts[vv * 3 + 2];
        const float rn = rsqrtf(vx * vx + vy * vy + vz * vz);
        const float ang = (vx * vnormals[vv * 3 + 0] +
                           vy * vnormals[vv * 3 + 1] +
                           vz * vnormals[vv * 3 + 2]) * rn;
        const float var = 2.0f / 7.0f;
        const float t = (ang - (-1.0f + var * (float)jj)) / var;  // linspace(-1,1,8)
        row[136 + jj] = f2bf(__expf(-t * t));
    } else if (j < 40) {
        ((unsigned int*)row)[72 + (j - 32)] = 0u;  // cols 144..159 = 0
    }
}

// msg = sigmoid(bn(filter)) * softplus(bn(core)); scatter-add into agg[V,128].
__global__ __launch_bounds__(256)
void msg_scatter(const unsigned short* __restrict__ Z2,  // [E,256] bf16
                 const float* __restrict__ sc, const float* __restrict__ sh,
                 const int* __restrict__ vids,
                 float* __restrict__ agg, int E)
{
    const long long i = (long long)blockIdx.x * 256 + threadIdx.x;
    if (i >= (long long)E * 128) return;
    const int e = (int)(i >> 7);
    const int h = (int)(i & 127);
    const unsigned short* r = Z2 + (long long)e * 256;
    const float f = bf2f(r[h]) * sc[h] + sh[h];
    const float c = bf2f(r[128 + h]) * sc[128 + h] + sh[128 + h];
    const float sig = 1.0f / (1.0f + __expf(-f));
    const float sp  = (c > 20.0f) ? c : __logf(1.0f + __expf(c));
    atomicAdd(&agg[(long long)vids[e] * 128 + h], sig * sp);
}

// Xcat[v] = [ bf16(agg[v]) , bf16(bn_geom2(Z2_geom[v])) ]
__global__ __launch_bounds__(256)
void concat_feat(const float* __restrict__ agg,
                 const unsigned short* __restrict__ Zg,
                 const float* __restrict__ sc, const float* __restrict__ sh,
                 unsigned short* __restrict__ Xcat, int V)
{
    const long long i = (long long)blockIdx.x * 256 + threadIdx.x;
    if (i >= (long long)V * 256) return;
    const int v = (int)(i >> 8);
    const int c = (int)(i & 255);
    unsigned short r;
    if (c < 128) {
        r = f2bf(agg[(long long)v * 128 + c]);
    } else {
        const int h = c - 128;
        r = f2bf(bf2f(Zg[(long long)v * 128 + h]) * sc[h] + sh[h]);
    }
    Xcat[i] = r;
}

// out[m,128] (f32) = bn(Z[m,128])
__global__ __launch_bounds__(256)
void bn_out(const unsigned short* __restrict__ Z,
            const float* __restrict__ sc, const float* __restrict__ sh,
            float* __restrict__ out, long long M)
{
    const long long i = (long long)blockIdx.x * 256 + threadIdx.x;
    if (i >= M * 128) return;
    const int h = (int)(i & 127);
    out[i] = bf2f(Z[i]) * sc[h] + sh[h];
}

// ---------------------------------------------------------------------------
// Host orchestration
// ---------------------------------------------------------------------------
extern "C" void kernel_launch(void* const* d_in, const int* in_sizes, int n_in,
                              void* d_out, int out_size, void* d_ws, size_t ws_size,
                              hipStream_t stream)
{
    (void)in_sizes; (void)n_in; (void)out_size; (void)ws_size;

    constexpr int E  = 800000;
    constexpr int V  = 100000;
    constexpr int NN = 20000;

    // setup_inputs() dict order, param tuples flattened in place (8 leaves each)
    const float* graph_x   = (const float*)d_in[0];
    const float* surface_x = (const float*)d_in[1];
    const float* node_pos  = (const float*)d_in[2];
    const float* verts     = (const float*)d_in[3];
    const float* vnormals  = (const float*)d_in[4];
    const float* vdist     = (const float*)d_in[5];
    const int CH = 6, SU = 14, GE = 22, FE = 30;
    auto P = [&](int base, int k) { return (const float*)d_in[base + k]; };
    // tuple: 0=W1 1=b1 2=g1 3=be1 4=W2 5=b2 6=g2 7=be2 (biases cancel under BN)
    const int* nbr_ind  = (const int*)d_in[38];
    const int* nbr_vids = (const int*)d_in[39];

    // ---- workspace bump allocator ----
    char* wsb = (char*)d_ws;
    size_t off = 0;
    auto alloc = [&](size_t bytes) -> void* {
        off = (off + 255) & ~(size_t)255;
        void* p = wsb + off;
        off += bytes;
        return p;
    };
    typedef unsigned short u16;
    auto wfsz = [](int K, int N) { return (size_t)N * (((K + 31) / 32) * 32) * 2; };
    u16* wb_c1 = (u16*)alloc(wfsz(128, 128));
    u16* wb_c2 = (u16*)alloc(wfsz(128, 128));
    u16* wb_s1 = (u16*)alloc(wfsz(144, 128));
    u16* wb_s2 = (u16*)alloc(wfsz(128, 256));
    u16* wb_g1 = (u16*)alloc(wfsz(64, 128));
    u16* wb_g2 = (u16*)alloc(wfsz(128, 128));
    u16* wb_f1 = (u16*)alloc(wfsz(256, 128));
    u16* wb_f2 = (u16*)alloc(wfsz(128, 128));
    u16* gx_bf = (u16*)alloc((size_t)NN * 128 * 2);
    u16* sx_bf = (u16*)alloc((size_t)V * 64 * 2);
    u16* Xe    = (u16*)alloc((size_t)E * XE_LD * 2);
    u16* Z1s   = (u16*)alloc((size_t)E * 128 * 2);
    u16* Z2s   = (u16*)alloc((size_t)E * 256 * 2);
    u16* Z1g   = (u16*)alloc((size_t)V * 128 * 2);
    u16* Z2g   = (u16*)alloc((size_t)V * 128 * 2);
    u16* Z1c   = (u16*)alloc((size_t)NN * 128 * 2);
    u16* Z2c   = (u16*)alloc((size_t)NN * 128 * 2);
    float* agg = (float*)alloc((size_t)V * 128 * 4);
    u16* Xcat  = (u16*)alloc((size_t)V * 256 * 2);
    u16* Z1f   = (u16*)alloc((size_t)V * 128 * 2);
    u16* Z2f   = (u16*)alloc((size_t)V * 128 * 2);
    float* stats = (float*)alloc((size_t)8 * 1024 * 4);
    auto SUM = [&](int s) { return stats + s * 1024; };
    auto SQ  = [&](int s) { return stats + s * 1024 + 256; };
    auto SC  = [&](int s) { return stats + s * 1024 + 512; };
    auto SH  = [&](int s) { return stats + s * 1024 + 768; };

    // ---- zero accumulators (graph-capturable async memsets) ----
    hipMemsetAsync(stats, 0, (size_t)8 * 1024 * 4, stream);
    hipMemsetAsync(agg,   0, (size_t)V * 128 * 4, stream);

    // ---- pack weights into B-fragment layout ----
    auto packw = [&](const float* src, u16* dst, int K, int N) {
        const int kt = (K + 31) / 32;
        const long long total = (long long)(N >> 4) * kt * 512;
        pack_w<<<(int)((total + 255) / 256), 256, 0, stream>>>(src, dst, K, N, kt);
    };
    packw(P(CH, 0), wb_c1, 128, 128);
    packw(P(CH, 4), wb_c2, 128, 128);
    packw(P(SU, 0), wb_s1, 144, 128);   // ktiles=5, zero-padded to K=160
    packw(P(SU, 4), wb_s2, 128, 256);
    packw(P(GE, 0), wb_g1, 64, 128);
    packw(P(GE, 4), wb_g2, 128, 128);
    packw(P(FE, 0), wb_f1, 256, 128);
    packw(P(FE, 4), wb_f2, 128, 128);

    // ---- raw inputs -> bf16 ----
    {
        const long long n4 = (long long)NN * 128 / 4;
        conv_f32_bf16_v4<<<(int)((n4 + 255) / 256), 256, 0, stream>>>(graph_x, gx_bf, n4);
    }
    {
        const long long n4 = (long long)V * 64 / 4;
        conv_f32_bf16_v4<<<(int)((n4 + 255) / 256), 256, 0, stream>>>(surface_x, sx_bf, n4);
    }

    // ---- edge features [E,160] ----
    edge_feats<<<(E + 3) / 4, 256, 0, stream>>>(gx_bf, node_pos, verts, vnormals,
                                                vdist, nbr_ind, nbr_vids, Xe, E);

    auto gemm = [&](const u16* A, const float* asc, const float* ash,
                    const u16* W, u16* Z, int stage, int M, int K, int N) {
        dim3 grid((M + TM - 1) / TM, N / TN);
        if (asc)
            gemm_bf16_bn<true><<<grid, 256, 0, stream>>>(A, asc, ash, W, Z,
                                                         SUM(stage), SQ(stage), M, K, N);
        else
            gemm_bf16_bn<false><<<grid, 256, 0, stream>>>(A, asc, ash, W, Z,
                                                          SUM(stage), SQ(stage), M, K, N);
    };
    auto fin = [&](int stage, const float* g, const float* b, int N, int M) {
        bn_finalize<<<1, 256, 0, stream>>>(SUM(stage), SQ(stage), g, b,
                                           SC(stage), SH(stage), N, 1.0f / (float)M);
    };

    // ---- surf MLP: [E,160(=144+pad)] -> 128 -> 256 ----
    gemm(Xe, nullptr, nullptr, wb_s1, Z1s, 0, E, XE_LD, 128);
    fin(0, P(SU, 2), P(SU, 3), 128, E);
    gemm(Z1s, SC(0), SH(0), wb_s2, Z2s, 1, E, 128, 256);
    fin(1, P(SU, 6), P(SU, 7), 256, E);

    // ---- geom MLP: [V,64] -> 128 -> 128 ----
    gemm(sx_bf, nullptr, nullptr, wb_g1, Z1g, 2, V, 64, 128);
    fin(2, P(GE, 2), P(GE, 3), 128, V);
    gemm(Z1g, SC(2), SH(2), wb_g2, Z2g, 3, V, 128, 128);
    fin(3, P(GE, 6), P(GE, 7), 128, V);

    // ---- chem MLP: [NN,128] -> 128 -> 128 ----
    gemm(gx_bf, nullptr, nullptr, wb_c1, Z1c, 4, NN, 128, 128);
    fin(4, P(CH, 2), P(CH, 3), 128, NN);
    gemm(Z1c, SC(4), SH(4), wb_c2, Z2c, 5, NN, 128, 128);
    fin(5, P(CH, 6), P(CH, 7), 128, NN);

    // ---- messages + segment sum into agg[V,128] ----
    {
        const long long total = (long long)E * 128;
        msg_scatter<<<(int)((total + 255) / 256), 256, 0, stream>>>(
            Z2s, SC(1), SH(1), nbr_vids, agg, E);
    }

    // ---- Xcat = [agg | bn_geom2(Z2g)] ----
    {
        const long long total = (long long)V * 256;
        concat_feat<<<(int)((total + 255) / 256), 256, 0, stream>>>(
            agg, Z2g, SC(3), SH(3), Xcat, V);
    }

    // ---- feat MLP: [V,256] -> 128 -> 128 ----
    gemm(Xcat, nullptr, nullptr, wb_f1, Z1f, 6, V, 256, 128);
    fin(6, P(FE, 2), P(FE, 3), 128, V);
    gemm(Z1f, SC(6), SH(6), wb_f2, Z2f, 7, V, 128, 128);
    fin(7, P(FE, 6), P(FE, 7), 128, V);

    // ---- outputs: h_geom [V,128] then chem_feats_graph [NN,128] ----
    float* out = (float*)d_out;
    {
        const long long total = (long long)V * 128;
        bn_out<<<(int)((total + 255) / 256), 256, 0, stream>>>(
            Z2f, SC(7), SH(7), out, V);
    }
    {
        const long long total = (long long)NN * 128;
        bn_out<<<(int)((total + 255) / 256), 256, 0, stream>>>(
            Z2c, SC(5), SH(5), out + (long long)V * 128, NN);
    }
}
